// GIN_34789235098229
// MI455X (gfx1250) — compile-verified, matrix operand order checked by reading, not run on hardware
//
#include <hip/hip_runtime.h>

#define D 64
#define GNUM 512
#define GIN_BN_EPS 1e-5f

typedef __attribute__((ext_vector_type(2)))  float    v2f;
typedef __attribute__((ext_vector_type(8)))  float    v8f;
typedef __attribute__((ext_vector_type(16))) _Float16 v16h;

// Prefer exact fp32 WMMA (matches the fp32 reference); fall back to the
// codegen-confirmed f16 WMMA (f32 accumulate) if the f32 builtin is absent.
#if __has_builtin(__builtin_amdgcn_wmma_f32_16x16x4_f32)
#define USE_F32_WMMA 1
#else
#define USE_F32_WMMA 0
#endif

// ---------------------------------------------------------------------------
// degree / graph-count helpers
// ---------------------------------------------------------------------------
__global__ void gin_count(const int* __restrict__ idx, float* __restrict__ cnt, int n)
{
    int i = blockIdx.x * blockDim.x + threadIdx.x;
    if (i < n) atomicAdd(&cnt[idx[i]], 1.0f);
}

__global__ void gin_invert(float* __restrict__ v, int n)
{
    int i = blockIdx.x * blockDim.x + threadIdx.x;
    if (i < n) v[i] = (v[i] > 0.0f) ? (1.0f / v[i]) : 0.0f;
}

// ---------------------------------------------------------------------------
// edge scatter: agg[dst] += h[src]   (16 threads per edge, float4 granularity)
// ---------------------------------------------------------------------------
__global__ void gin_scatter(const float* __restrict__ h, const int* __restrict__ src,
                            const int* __restrict__ dst, float* __restrict__ agg, int e)
{
    long long idx = (long long)blockIdx.x * blockDim.x + threadIdx.x;
    int edge = (int)(idx >> 4);
    if (edge >= e) return;
    int q = ((int)idx & 15) * 4;
    int s = src[edge];
    int d = dst[edge];
    float4 v = *(const float4*)(h + (size_t)s * D + q);
    float* ap = agg + (size_t)d * D + q;
    atomicAdd(ap + 0, v.x);
    atomicAdd(ap + 1, v.y);
    atomicAdd(ap + 2, v.z);
    atomicAdd(ap + 3, v.w);
}

// ---------------------------------------------------------------------------
// 16x64 row-tile GEMM against a 64x64 weight panel (transposed, pitch 65, LDS)
// acc[t] = relu-ready f32 accumulators for output cols [16t, 16t+16)
// ---------------------------------------------------------------------------
__device__ __forceinline__ void gin_tile_gemm(const float* asrc, const float* wt,
                                              const float* bias, int lane, v8f acc[4])
{
    const int col0 = lane & 15;
#if USE_F32_WMMA
    // A layout (f32 16x16x4): lane<16 -> K pair {4s,4s+1}, lane>=16 -> {4s+2,4s+3}
    const int koff = (lane < 16) ? 0 : 2;
    float a0[16], a1[16];
#pragma unroll
    for (int s = 0; s < 16; ++s) {
        a0[s] = asrc[4 * s + koff];
        a1[s] = asrc[4 * s + koff + 1];
    }
#pragma unroll
    for (int t = 0; t < 4; ++t) {
        const int col = t * 16 + col0;
        const float bv = bias[col];
        v8f c;
#pragma unroll
        for (int r = 0; r < 8; ++r) c[r] = bv;
        const float* wc = wt + col * 65;
#pragma unroll
        for (int s = 0; s < 16; ++s) {
            v2f a, b;
            a[0] = a0[s];             a[1] = a1[s];
            b[0] = wc[4 * s + koff];  b[1] = wc[4 * s + koff + 1];
            c = __builtin_amdgcn_wmma_f32_16x16x4_f32(false, a, false, b,
                                                      (short)0, c, false, false);
        }
        acc[t] = c;
    }
#else
    // A layout (f16 16x16x32): halves 0..7 -> K=kb0.., halves 8..15 -> K=kb1..
    const int kb0 = (lane < 16) ? 0 : 8;
    const int kb1 = kb0 + 16;
    v16h alo, ahi;
#pragma unroll
    for (int i = 0; i < 8; ++i) {
        alo[i]     = (_Float16)asrc[kb0 + i];
        alo[8 + i] = (_Float16)asrc[kb1 + i];
        ahi[i]     = (_Float16)asrc[32 + kb0 + i];
        ahi[8 + i] = (_Float16)asrc[32 + kb1 + i];
    }
#pragma unroll
    for (int t = 0; t < 4; ++t) {
        const int col = t * 16 + col0;
        const float bv = bias[col];
        v8f c;
#pragma unroll
        for (int r = 0; r < 8; ++r) c[r] = bv;
        const float* wc = wt + col * 65;
        v16h blo, bhi;
#pragma unroll
        for (int i = 0; i < 8; ++i) {
            blo[i]     = (_Float16)wc[kb0 + i];
            blo[8 + i] = (_Float16)wc[kb1 + i];
            bhi[i]     = (_Float16)wc[32 + kb0 + i];
            bhi[8 + i] = (_Float16)wc[32 + kb1 + i];
        }
        c = __builtin_amdgcn_wmma_f32_16x16x32_f16(false, alo, false, blo,
                                                   (short)0, c, false, false);
        c = __builtin_amdgcn_wmma_f32_16x16x32_f16(false, ahi, false, bhi,
                                                   (short)0, c, false, false);
        acc[t] = c;
    }
#endif
}

// ---------------------------------------------------------------------------
// fused  z = h + agg*invdeg ;  z2 = relu( relu(z@W1+b1) @ W2 + b2 )
// 256 threads = 8 waves, each wave owns a 16x64 row tile (128 rows per block)
// ---------------------------------------------------------------------------
__global__ void __launch_bounds__(256)
gin_mlp(const float* __restrict__ hin, const float* __restrict__ agg,
        const float* __restrict__ invdeg,
        const float* __restrict__ W1, const float* __restrict__ b1,
        const float* __restrict__ W2, const float* __restrict__ b2,
        float* __restrict__ z2, int n)
{
    __shared__ float w1t[64 * 65];   // w1t[nout*65 + kin] = W1[kin*64 + nout]
    __shared__ float w2t[64 * 65];
    __shared__ float zls[8][1024];   // per-wave 16x64 activation scratch
    __shared__ float bias[128];

    const int tid  = threadIdx.x;
    const int wave = tid >> 5;
    const int lane = tid & 31;
    const int row0 = blockIdx.x * 128 + wave * 16;

    for (int i = tid; i < 4096; i += 256) {
        const int kk = i >> 6, nn = i & 63;
        w1t[nn * 65 + kk] = W1[i];
        w2t[nn * 65 + kk] = W2[i];
    }
    if (tid < 64) { bias[tid] = b1[tid]; bias[64 + tid] = b2[tid]; }

    // fused aggregation-add while staging the 16x64 tile into LDS
    float* zw = zls[wave];
#pragma unroll
    for (int j = 0; j < 8; ++j) {
        const int e0 = (j * 32 + lane) * 4;
        int grow = row0 + (e0 >> 6);
        if (grow > n - 1) grow = n - 1;          // clamp (EXEC must stay full)
        const int k = e0 & 63;
        const float4 hv = *(const float4*)(hin + (size_t)grow * D + k);
        const float4 av = *(const float4*)(agg + (size_t)grow * D + k);
        const float id = invdeg[grow];
        float4 zv;
        zv.x = hv.x + av.x * id;
        zv.y = hv.y + av.y * id;
        zv.z = hv.z + av.z * id;
        zv.w = hv.w + av.w * id;
        *(float4*)(zw + e0) = zv;
    }
    __syncthreads();

    v8f acc[4];
    gin_tile_gemm(zw + (lane & 15) * 64, w1t, bias, lane, acc);

    __syncthreads();
#pragma unroll
    for (int t = 0; t < 4; ++t)
#pragma unroll
        for (int r = 0; r < 8; ++r) {
            const int M = (lane < 16) ? r : r + 8;
            zw[M * 64 + t * 16 + (lane & 15)] = fmaxf(acc[t][r], 0.0f);
        }
    __syncthreads();

    gin_tile_gemm(zw + (lane & 15) * 64, w2t, bias + 64, lane, acc);

    const int col0 = lane & 15;
#pragma unroll
    for (int t = 0; t < 4; ++t)
#pragma unroll
        for (int r = 0; r < 8; ++r) {
            const int M = (lane < 16) ? r : r + 8;
            const int row = row0 + M;
            if (row < n)
                z2[(size_t)row * D + t * 16 + col0] = fmaxf(acc[t][r], 0.0f);
        }
}

// ---------------------------------------------------------------------------
// BatchNorm statistics (sum, sum of squares per feature)
// ---------------------------------------------------------------------------
__global__ void gin_bn_stats(const float* __restrict__ z, float* __restrict__ sums, int n)
{
    __shared__ float s1[256];
    __shared__ float s2[256];
    const int tid = threadIdx.x;
    const int col = tid & (D - 1);
    const int rg  = tid >> 6;
    float a = 0.0f, b = 0.0f;
    for (int r = blockIdx.x * 4 + rg; r < n; r += gridDim.x * 4) {
        const float v = z[(size_t)r * D + col];
        a += v; b += v * v;
    }
    s1[tid] = a; s2[tid] = b;
    __syncthreads();
    if (tid < D) {
        a = s1[tid] + s1[tid + 64] + s1[tid + 128] + s1[tid + 192];
        b = s2[tid] + s2[tid + 64] + s2[tid + 128] + s2[tid + 192];
        atomicAdd(&sums[tid], a);
        atomicAdd(&sums[D + tid], b);
    }
}

__global__ void gin_bn_coef(const float* __restrict__ sums, const float* __restrict__ gamma,
                            const float* __restrict__ beta, float* __restrict__ coef, int n)
{
    const int d = threadIdx.x;
    const float inv_n = 1.0f / (float)n;
    const float mu  = sums[d] * inv_n;
    const float var = sums[D + d] * inv_n - mu * mu;
    const float rs  = rsqrtf(var + GIN_BN_EPS);
    const float A   = rs * gamma[d];
    coef[d]     = A;
    coef[D + d] = beta[d] - mu * A;
}

// h = z*A + B ; node_pool += h ; g_pool[batch] += h * inv_gcnt[batch]
__global__ void gin_bn_apply(const float* __restrict__ z, const float* __restrict__ coef,
                             const int* __restrict__ batch, const float* __restrict__ invg,
                             float* __restrict__ hout, float* __restrict__ npool,
                             float* __restrict__ gpool, int n)
{
    const int i = blockIdx.x * blockDim.x + threadIdx.x;
    if (i >= n * D) return;
    const int r = i >> 6, d = i & 63;
    const float hv = z[i] * coef[d] + coef[D + d];
    hout[i] = hv;
    npool[i] += hv;
    const int g = batch[r];
    atomicAdd(&gpool[(size_t)g * D + d], hv * invg[g]);
}

// ---------------------------------------------------------------------------
extern "C" void kernel_launch(void* const* d_in, const int* in_sizes, int n_in,
                              void* d_out, int out_size, void* d_ws, size_t ws_size,
                              hipStream_t stream)
{
    const float* x     = (const float*)d_in[0];
    const float* W1    = (const float*)d_in[1];
    const float* b1    = (const float*)d_in[2];
    const float* W2    = (const float*)d_in[3];
    const float* b2    = (const float*)d_in[4];
    const float* gamma = (const float*)d_in[5];
    const float* beta  = (const float*)d_in[6];
    const int*   eidx  = (const int*)d_in[7];
    const int*   batch = (const int*)d_in[8];

    const int n = in_sizes[0] / D;
    const int e = in_sizes[7] / 2;
    const int* src = eidx;
    const int* dst = eidx + e;

    float* npool = (float*)d_out;              // N*D
    float* gpool = npool + (size_t)n * D;      // G*D

    float* hbuf   = (float*)d_ws;              // N*D
    float* agg    = hbuf  + (size_t)n * D;     // N*D
    float* z2     = agg   + (size_t)n * D;     // N*D
    float* invdeg = z2    + (size_t)n * D;     // N (deg accum, then inverted)
    float* invg   = invdeg + n;                // GNUM
    float* bnsum  = invg  + GNUM;              // 2*D
    float* coef   = bnsum + 2 * D;             // 2*D

    hipMemsetAsync(d_out, 0, (size_t)out_size * sizeof(float), stream);
    hipMemsetAsync(invdeg, 0, (size_t)(n + GNUM) * sizeof(float), stream);

    gin_count<<<(e + 255) / 256, 256, 0, stream>>>(dst, invdeg, e);
    gin_count<<<(n + 255) / 256, 256, 0, stream>>>(batch, invg, n);
    gin_invert<<<(n + 255) / 256, 256, 0, stream>>>(invdeg, n);
    gin_invert<<<(GNUM + 255) / 256, 256, 0, stream>>>(invg, GNUM);

    const float* hin = x;
    const int mlp_blocks = (n + 127) / 128;
    const long long sth = (long long)e * 16;

    for (int l = 0; l < 4; ++l) {
        hipMemsetAsync(agg, 0, (size_t)n * D * sizeof(float), stream);
        hipMemsetAsync(bnsum, 0, 2 * D * sizeof(float), stream);

        gin_scatter<<<(unsigned)((sth + 255) / 256), 256, 0, stream>>>(hin, src, dst, agg, e);
        gin_mlp<<<mlp_blocks, 256, 0, stream>>>(hin, agg, invdeg,
                                                W1 + (size_t)l * D * D, b1 + l * D,
                                                W2 + (size_t)l * D * D, b2 + l * D,
                                                z2, n);
        gin_bn_stats<<<512, 256, 0, stream>>>(z2, bnsum, n);
        gin_bn_coef<<<1, D, 0, stream>>>(bnsum, gamma + l * D, beta + l * D, coef, n);
        gin_bn_apply<<<((n * D) + 255) / 256, 256, 0, stream>>>(z2, coef, batch, invg,
                                                                hbuf, npool, gpool, n);
        hin = hbuf;
    }
}
